// Linear_reparam_gaussian_12876311953605
// MI455X (gfx1250) — matrix operand order repaired and checked by reading, not compiled, not run on hardware
//
#include <hip/hip_runtime.h>
#include <hip/hip_bf16.h>

// ---------- vector types matching the gfx1250 WMMA builtin signatures ----------
typedef __attribute__((ext_vector_type(16))) __bf16 v16bf;
typedef __attribute__((ext_vector_type(8)))  __bf16 bf16x8;
typedef __attribute__((ext_vector_type(4)))  __bf16 bf16x4;
typedef __attribute__((ext_vector_type(8)))  float  v8f;
typedef __attribute__((ext_vector_type(4)))  float  f32x4;

// GCC-style vector for the async-LDS builtin (clang printed __vector_size__ int)
typedef int v4i __attribute__((vector_size(16)));
typedef __attribute__((address_space(1))) v4i global_v4i;
typedef __attribute__((address_space(3))) v4i lds_v4i;

#define N_ROWS   4096
#define K_DIM    2048
#define OUT_DIM  2048

#define BM 128           // block rows (of input)
#define BN 128           // block cols (of output)
#define BK 32            // K tile
#define KTILES (K_DIM / BK)
#define PAD_K 40         // 32 + 8 bf16 pad: 80B row stride -> 16 lanes hit 16 distinct banks
#define A_TILE_ELEMS (BM * PAD_K)          // 5120 bf16
#define B_TILE_ELEMS (BN * PAD_K)          // 5120 bf16
#define STAGE_ELEMS  (A_TILE_ELEMS + B_TILE_ELEMS)   // 10240 bf16 = 20 KiB; x2 stages = 40 KiB

// ---------------- CDNA5 async global->LDS path (guarded) ----------------
#if __has_builtin(__builtin_amdgcn_global_load_async_to_lds_b128)
#define HAVE_ASYNC_LDS 1
#endif

__device__ __forceinline__ void async_cp16(const __bf16* g, __bf16* l) {
#ifdef HAVE_ASYNC_LDS
    __builtin_amdgcn_global_load_async_to_lds_b128(
        (global_v4i*)g, (lds_v4i*)l, /*offset=*/0, /*cpol=*/0);
#else
    *(bf16x8*)l = *(const bf16x8*)g;   // fallback: global->VGPR->ds_store
#endif
}

__device__ __forceinline__ void wait_async_lds() {
#ifdef HAVE_ASYNC_LDS
#if __has_builtin(__builtin_amdgcn_s_wait_asynccnt)
    __builtin_amdgcn_s_wait_asynccnt(0);
#else
    asm volatile("s_wait_asynccnt 0x0" ::: "memory");
#endif
#endif
}

__device__ __forceinline__ float softplus_f(float x) {
    return (x > 20.0f) ? x : log1pf(__expf(x));
}

// ---------------- prep 1: f32 input -> bf16 (4 elems / thread) ----------------
__global__ __launch_bounds__(256) void cvt_input_bf16(const float* __restrict__ in,
                                                      __bf16* __restrict__ out) {
    int idx = (blockIdx.x * 256 + threadIdx.x) * 4;
    f32x4 v = *(const f32x4*)(in + idx);
    bf16x4 o;
#pragma unroll
    for (int i = 0; i < 4; ++i) o[i] = (__bf16)v[i];
    *(bf16x4*)(out + idx) = o;
}

// ------- prep 2: w = mu + softplus(rho) * eps  -> bf16 (4 elems / thread) -------
__global__ __launch_bounds__(256) void prep_weight_bf16(const float* __restrict__ mu,
                                                        const float* __restrict__ rho,
                                                        const float* __restrict__ eps,
                                                        __bf16* __restrict__ wbf) {
    int idx = (blockIdx.x * 256 + threadIdx.x) * 4;
    f32x4 m = *(const f32x4*)(mu  + idx);
    f32x4 r = *(const f32x4*)(rho + idx);
    f32x4 e = *(const f32x4*)(eps + idx);
    bf16x4 w;
#pragma unroll
    for (int i = 0; i < 4; ++i) w[i] = (__bf16)(m[i] + softplus_f(r[i]) * e[i]);
    *(bf16x4*)(wbf + idx) = w;
}

// ------- cooperative async staging: 128x32 A-tile + 128x32 B-tile -> LDS -------
// 128 threads; each thread issues 8 x b128 async copies (16B granularity).
__device__ __forceinline__ void stage_tiles(const __bf16* __restrict__ A,
                                            const __bf16* __restrict__ W,
                                            __bf16* lds,
                                            int blockRow, int blockCol,
                                            int k0, int tid) {
#pragma unroll
    for (int i = 0; i < 4; ++i) {
        const int c   = tid + i * 128;     // 0..511
        const int row = c >> 2;            // 0..127
        const int col = (c & 3) * 8;       // 0,8,16,24 (bf16 elems; 16B chunks)
        async_cp16(A + (size_t)(blockRow + row) * K_DIM + k0 + col,
                   lds + row * PAD_K + col);
    }
    __bf16* ldsB = lds + A_TILE_ELEMS;
#pragma unroll
    for (int i = 0; i < 4; ++i) {
        const int c   = tid + i * 128;
        const int row = c >> 2;
        const int col = (c & 3) * 8;
        async_cp16(W + (size_t)(blockCol + row) * K_DIM + k0 + col,
                   ldsB + row * PAD_K + col);
    }
}

// ---------------- GEMM: C[N,OUT] = A[N,K] * W[OUT,K]^T + bias ----------------
// 128 threads = 4 waves (2x2). Block tile 128x128, K-tile 32, double-buffered LDS.
// Each wave computes a 64x64 sub-tile = 4x4 WMMA fragments (16 wmma : 16 ds_b128).
__global__ __launch_bounds__(128) void gemm_wmma_bf16(const __bf16* __restrict__ A,
                                                      const __bf16* __restrict__ W,
                                                      const float*  __restrict__ bias,
                                                      float* __restrict__ C) {
    __shared__ __bf16 smem[2][STAGE_ELEMS];

    const int tid  = threadIdx.x;
    const int lane = tid & 31;
    const int wave = tid >> 5;                 // 0..3
    const int waveRow = (wave >> 1) * 64;      // 0,64
    const int waveCol = (wave &  1) * 64;      // 0,64

    const int blockRow = blockIdx.y * BM;
    const int blockCol = blockIdx.x * BN;

    const int m  = lane & 15;
    const int hi = lane >> 4;                  // lane half selects K sub-chunk

    v8f acc[4][4];
#pragma unroll
    for (int i = 0; i < 4; ++i)
#pragma unroll
        for (int j = 0; j < 4; ++j) acc[i][j] = (v8f){};

    // prologue: prefetch stage 0
    stage_tiles(A, W, &smem[0][0], blockRow, blockCol, 0, tid);

    for (int kt = 0; kt < KTILES; ++kt) {
        wait_async_lds();      // my async copies for smem[kt&1] have landed
        __syncthreads();       // everyone's copies landed; everyone done reading smem[(kt+1)&1]

        if (kt + 1 < KTILES)   // overlap next stage fill with this stage's WMMAs
            stage_tiles(A, W, &smem[(kt + 1) & 1][0], blockRow, blockCol,
                        (kt + 1) * BK, tid);

        const __bf16* bufA = &smem[kt & 1][0];
        const __bf16* bufB = bufA + A_TILE_ELEMS;
        // A 16x32 bf16 frag: lanes 0-15 hold K[0:8)+K[16:24); lanes 16-31 K[8:16)+K[24:32)
        const __bf16* sAf = bufA + (waveRow + m) * PAD_K + hi * 8;
        // B 32x16 bf16 frag: lanes 0-15 hold K[0:8)+K[16:24); lanes 16-31 K[8:16)+K[24:32); N = lane&15
        const __bf16* sBf = bufB + (waveCol + m) * PAD_K + hi * 8;

        v16bf af[4], bfr[4];
        union Pun { v16bf v; bf16x8 h[2]; };
#pragma unroll
        for (int i = 0; i < 4; ++i) {
            const __bf16* p = sAf + i * 16 * PAD_K;
            Pun u;
            u.h[0] = *(const bf16x8*)(p);
            u.h[1] = *(const bf16x8*)(p + 16);
            af[i] = u.v;
        }
#pragma unroll
        for (int j = 0; j < 4; ++j) {
            const __bf16* p = sBf + j * 16 * PAD_K;
            Pun u;
            u.h[0] = *(const bf16x8*)(p);
            u.h[1] = *(const bf16x8*)(p + 16);
            bfr[j] = u.v;
        }

#pragma unroll
        for (int i = 0; i < 4; ++i)
#pragma unroll
            for (int j = 0; j < 4; ++j)
                acc[i][j] = __builtin_amdgcn_wmma_f32_16x16x32_bf16(
                    false, af[i], false, bfr[j],
                    (short)0, acc[i][j], false, false);
    }

    // --- epilogue: D layout: VGPR r -> M = r + 8*hi ; N = lane&15 ---
    const int col0 = blockCol + waveCol + m;
    const int row0 = blockRow + waveRow + hi * 8;
#pragma unroll
    for (int i = 0; i < 4; ++i) {
#pragma unroll
        for (int j = 0; j < 4; ++j) {
            const int col = col0 + j * 16;
            const float b = bias[col];
            float* cptr = C + (size_t)(row0 + i * 16) * OUT_DIM + col;
#pragma unroll
            for (int r = 0; r < 8; ++r)
                cptr[(size_t)r * OUT_DIM] = acc[i][j][r] + b;
        }
    }
}

extern "C" void kernel_launch(void* const* d_in, const int* in_sizes, int n_in,
                              void* d_out, int out_size, void* d_ws, size_t ws_size,
                              hipStream_t stream) {
    const float* input      = (const float*)d_in[0];  // [4096,2048]
    const float* weight_mu  = (const float*)d_in[1];  // [2048,2048]
    const float* weight_rho = (const float*)d_in[2];  // [2048,2048]
    const float* eps_w      = (const float*)d_in[3];  // [2048,2048]
    const float* bias       = (const float*)d_in[4];  // [2048]
    float* out = (float*)d_out;                       // [4096,2048]

    // workspace layout: bf16 A, then bf16 W (both L2-resident: 24 MiB total)
    __bf16* A_bf = (__bf16*)d_ws;
    __bf16* W_bf = (__bf16*)((char*)d_ws + (size_t)N_ROWS * K_DIM * 2);

    cvt_input_bf16<<<(N_ROWS * K_DIM) / (256 * 4), 256, 0, stream>>>(input, A_bf);
    prep_weight_bf16<<<(OUT_DIM * K_DIM) / (256 * 4), 256, 0, stream>>>(
        weight_mu, weight_rho, eps_w, W_bf);

    dim3 grid(OUT_DIM / BN, N_ROWS / BM);   // (16, 32) = 512 blocks
    gemm_wmma_bf16<<<grid, 128, 0, stream>>>(A_bf, W_bf, bias, out);
}
